// MultiHeadAttn_69423851372922
// MI455X (gfx1250) — matrix operand order repaired and checked by reading, not compiled, not run on hardware
//
#include <hip/hip_runtime.h>
#include <hip/hip_bf16.h>
#include <math.h>

// ---------------------------------------------------------------------------
// MI455X (gfx1250) multi-head attention, f16 WMMA pipeline, f32 accumulate.
// All fragment feeds are 128-bit loads (global_load_b128 / ds_load_b128).
//
// Fragment layout conventions (CDNA5 ISA 7.12.2, wave32):
//  A (16x32 f16):  lane holds row m = lane&15 (both 16-lane halves hold the
//                  same rows); element e -> K = 16*half + e (order used here,
//                             same as B; not the ISA table order).
//                  => two contiguous 8-element (16B) runs per lane.
//  B (32x16 f16):  lane holds col n = lane&15; element e -> K = e + 16*half (order used here).
//                  => 16 contiguous K values (32B) per lane if K-major.
//  C/D (16x16 f32): reg r, lane -> row = r + 8*half, col = lane&15.
// ---------------------------------------------------------------------------

typedef __attribute__((ext_vector_type(16))) _Float16 v16h;   // 32 B
typedef __attribute__((ext_vector_type(8)))  _Float16 v8h;    // 16 B
typedef __attribute__((ext_vector_type(8)))  float    v8f;
typedef __attribute__((ext_vector_type(4)))  float    v4f;

union V16U { v16h v; v8h h[2]; };

#define WMMA_F16(A, B, C) \
  __builtin_amdgcn_wmma_f32_16x16x32_f16(false, (A), false, (B), (short)0, (C), false, false)

#define D_MODEL 1024
#define N_HEADS 16
#define D_K     64
#define ROWS    4096          // B * L = 4 * 1024
#define HEAD_ELEMS ((size_t)N_HEADS * ROWS * D_K)   // one of Qh/Kh/Vh in halves

// ---------------------------------------------------------------------------
// Kernel 1: per-head QKV projections.  X[4096,1024] (f32) x W[h][1024,64]
// -> f16.  Q,K stored [h][row][dk]; V stored transposed [h][dv][row] so the
// attention P*V B-fragment is a contiguous 32B load.
// grid = (64 row-groups, 16 heads, 3 matrices), block = 128 (4 waves).
// ---------------------------------------------------------------------------
__global__ void __launch_bounds__(128)
qkv_proj_kernel(const float* __restrict__ q, const float* __restrict__ k,
                const float* __restrict__ v,
                const float* __restrict__ wq, const float* __restrict__ wk,
                const float* __restrict__ wv,
                _Float16* __restrict__ ws_qkv) {
  const int lane = threadIdx.x & 31;
  const int wave = threadIdx.x >> 5;
  const int half = lane >> 4;
  const int l16  = lane & 15;
  const int h    = blockIdx.y;
  const int z    = blockIdx.z;

  const float* X = (z == 0) ? q : (z == 1) ? k : v;
  const float* W = (z == 0) ? wq : (z == 1) ? wk : wv;
  _Float16* O = ws_qkv + (size_t)z * HEAD_ELEMS;

  const int rt  = blockIdx.x * 4 + wave;   // 0..255 row tiles of 16
  const int row = rt * 16 + l16;

  // Transposed weight slab: [col 0..63][k 0..31]; rows are 64 B, 32 B aligned.
  __shared__ __align__(32) _Float16 lds_wT[64][32];

  v8f acc[4] = {};

  for (int k0 = 0; k0 < D_MODEL; k0 += 32) {
    __syncthreads();
    // Global reads coalesced in c (fastest dim of W); LDS writes transposed.
    for (int i = threadIdx.x; i < 32 * 64; i += 128) {
      const int kk = i >> 6, cc = i & 63;
      lds_wT[cc][kk] = (_Float16)W[((size_t)h * D_MODEL + (k0 + kk)) * D_K + cc];
    }
    __syncthreads();

    // A fragment: 4x float4 loads + packed f32->f16 converts.
    const float* xr = X + (size_t)row * D_MODEL + k0 + 16 * half;
    if (k0 + 32 < D_MODEL) __builtin_prefetch(xr + 32, 0, 1);
    const v4f f0 = *(const v4f*)(xr);
    const v4f f1 = *(const v4f*)(xr + 4);
    const v4f f2 = *(const v4f*)(xr + 8);
    const v4f f3 = *(const v4f*)(xr + 12);
    v16h a;
#pragma unroll
    for (int j = 0; j < 4; ++j) {
      a[j]      = (_Float16)f0[j];
      a[j + 4]  = (_Float16)f1[j];
      a[j + 8]  = (_Float16)f2[j];
      a[j + 12] = (_Float16)f3[j];
    }

#pragma unroll
    for (int ct = 0; ct < 4; ++ct) {
      // B fragment: 16 contiguous K halves per lane (2x ds_load_b128).
      const v16h bf = *(const v16h*)&lds_wT[ct * 16 + l16][16 * half];
      acc[ct] = WMMA_F16(a, bf, acc[ct]);
    }
  }

#pragma unroll
  for (int ct = 0; ct < 4; ++ct)
#pragma unroll
    for (int r = 0; r < 8; ++r) {
      const int grow = rt * 16 + r + 8 * half;
      const int gcol = ct * 16 + l16;
      const size_t idx = (z == 2)
          ? ((size_t)h * D_K + gcol) * ROWS + grow        // V: [h][dv][row]
          : ((size_t)h * ROWS + grow) * D_K + gcol;       // Q,K: [h][row][dk]
      O[idx] = (_Float16)acc[ct][r];
    }
}

// ---------------------------------------------------------------------------
// Kernel 2: flash attention.  One wave per (query-tile16, head, batch).
// Online softmax over 32-key blocks; scores scaled by 1/sqrt(d_model)=1/32.
// ---------------------------------------------------------------------------
__global__ void __launch_bounds__(32)
attention_kernel(const _Float16* __restrict__ ws_qkv, _Float16* __restrict__ Ow) {
  const int lane = threadIdx.x;
  const int half = lane >> 4;
  const int l16  = lane & 15;
  const int qt   = blockIdx.x;   // 0..63
  const int h    = blockIdx.y;   // 0..15
  const int b    = blockIdx.z;   // 0..3

  const _Float16* Qh  = ws_qkv;                       // [h][row][dk]
  const _Float16* Kh  = ws_qkv + HEAD_ELEMS;          // [h][row][dk]
  const _Float16* Vt  = ws_qkv + 2 * HEAD_ELEMS;      // [h][dv][row]
  const size_t qk_base = ((size_t)h * ROWS + (size_t)b * 1024) * D_K;

  __shared__ __align__(32) _Float16 lds_p[16][32];    // P: C-layout -> A-layout hop

  // Q fragments (dk = 64 -> two 16x32 A fragments), 16B vector loads.
  v16h qf[2];
  {
    const _Float16* qr = Qh + qk_base + (size_t)(qt * 16 + l16) * D_K;
#pragma unroll
    for (int ks = 0; ks < 2; ++ks) {
      V16U u;
      u.h[0] = *(const v8h*)(qr + ks * 32 + 16 * half);
      u.h[1] = *(const v8h*)(qr + ks * 32 + 16 * half + 8);
      qf[ks] = u.v;
    }
  }

  float m[8], l[8];
#pragma unroll
  for (int r = 0; r < 8; ++r) { m[r] = -INFINITY; l[r] = 0.0f; }
  v8f oacc[4] = {};

  for (int kb = 0; kb < 32; ++kb) {        // 32 blocks of 32 keys
    v8f s0 = {}, s1 = {};
#pragma unroll
    for (int ks = 0; ks < 2; ++ks) {
      // B fragments of K^T: col = key, per-lane K-dim halves contiguous (32B).
      const _Float16* kr0 =
          Kh + qk_base + (size_t)(kb * 32 + l16) * D_K + ks * 32 + 16 * half;
      if (ks == 0) __builtin_prefetch(kr0 + 32 * D_K, 0, 1);
      const v16h b0 = *(const v16h*)(kr0);
      const v16h b1 = *(const v16h*)(kr0 + 16 * D_K);
      s0 = WMMA_F16(qf[ks], b0, s0);
      s1 = WMMA_F16(qf[ks], b1, s1);
    }

    const float sc = 0.03125f;             // 1/sqrt(1024)
#pragma unroll
    for (int r = 0; r < 8; ++r) {
      float x0 = s0[r] * sc, x1 = s1[r] * sc;
      float vmax = fmaxf(x0, x1);
#pragma unroll
      for (int off = 8; off; off >>= 1)
        vmax = fmaxf(vmax, __shfl_xor(vmax, off, 32));   // stays in 16-lane half
      const float mnew  = fmaxf(m[r], vmax);
      const float alpha = __expf(m[r] - mnew);
      const float p0 = __expf(x0 - mnew), p1 = __expf(x1 - mnew);
      float rs = p0 + p1;
#pragma unroll
      for (int off = 8; off; off >>= 1)
        rs += __shfl_xor(rs, off, 32);
      l[r] = l[r] * alpha + rs;
      m[r] = mnew;
      const int mr = r + 8 * half;
      lds_p[mr][l16]      = (_Float16)p0;
      lds_p[mr][16 + l16] = (_Float16)p1;
#pragma unroll
      for (int ct = 0; ct < 4; ++ct) oacc[ct][r] *= alpha;
    }
    __syncthreads();

    // Re-read P as an A fragment: two 16B ds_load_b128 per lane.
    V16U pu;
    pu.h[0] = *(const v8h*)&lds_p[l16][16 * half];
    pu.h[1] = *(const v8h*)&lds_p[l16][16 * half + 8];
    const v16h pf = pu.v;
    __syncthreads();

#pragma unroll
    for (int ct = 0; ct < 4; ++ct) {
      // B fragment of V (transposed layout): contiguous 32B per lane.
      const _Float16* vr = Vt + ((size_t)h * D_K + ct * 16 + l16) * ROWS
                         + (size_t)b * 1024 + kb * 32 + 16 * half;
      const v16h bv = *(const v16h*)(vr);
      oacc[ct] = WMMA_F16(pf, bv, oacc[ct]);
    }
  }

  // Normalize and scatter into [b][s][h*64 + dv] (token-major for out-proj).
#pragma unroll
  for (int ct = 0; ct < 4; ++ct)
#pragma unroll
    for (int r = 0; r < 8; ++r) {
      const int mr = r + 8 * half;
      const float o = oacc[ct][r] / l[r];
      const size_t idx = ((size_t)(b * 1024 + qt * 16 + mr)) * D_MODEL
                       + h * D_K + ct * 16 + l16;
      Ow[idx] = (_Float16)o;
    }
}

// ---------------------------------------------------------------------------
// Kernel 3: output projection Ow[4096,1024] @ w_proj^T + residual (q).
// grid = (64 row-groups, 16 col-strips of 64), block = 128 (4 waves).
// ---------------------------------------------------------------------------
__global__ void __launch_bounds__(128)
proj_residual_kernel(const _Float16* __restrict__ Ow,
                     const float* __restrict__ wproj,
                     const float* __restrict__ resid,
                     float* __restrict__ out) {
  const int lane = threadIdx.x & 31;
  const int wave = threadIdx.x >> 5;
  const int half = lane >> 4;
  const int l16  = lane & 15;
  const int rt     = blockIdx.x * 4 + wave;  // 0..255
  const int dstrip = blockIdx.y;             // 0..15
  const int row    = rt * 16 + l16;

  // Transposed slab of w_proj: [d 0..63][k 0..31].
  __shared__ __align__(32) _Float16 lds_wT[64][32];

  v8f acc[4] = {};

  for (int k0 = 0; k0 < D_MODEL; k0 += 32) {
    __syncthreads();
    // Global reads coalesced in k (fastest dim of w_proj row); LDS row writes.
    for (int i = threadIdx.x; i < 64 * 32; i += 128) {
      const int dd = i >> 5, kk = i & 31;
      lds_wT[dd][kk] = (_Float16)wproj[(size_t)(dstrip * 64 + dd) * D_MODEL + k0 + kk];
    }
    __syncthreads();

    // A fragment from Ow: two 16B vector loads.
    const _Float16* ar = Ow + (size_t)row * D_MODEL + k0 + 16 * half;
    if (k0 + 32 < D_MODEL) __builtin_prefetch(ar + 32, 0, 1);
    V16U au;
    au.h[0] = *(const v8h*)(ar);
    au.h[1] = *(const v8h*)(ar + 8);
    const v16h a = au.v;

#pragma unroll
    for (int ct = 0; ct < 4; ++ct) {
      const v16h bf = *(const v16h*)&lds_wT[ct * 16 + l16][16 * half];
      acc[ct] = WMMA_F16(a, bf, acc[ct]);
    }
  }

#pragma unroll
  for (int ct = 0; ct < 4; ++ct)
#pragma unroll
    for (int r = 0; r < 8; ++r) {
      const int grow = rt * 16 + r + 8 * half;
      const int gcol = dstrip * 64 + ct * 16 + l16;
      const size_t idx = (size_t)grow * D_MODEL + gcol;
      out[idx] = acc[ct][r] + resid[idx];
    }
}

// ---------------------------------------------------------------------------
// Kernel 4: in-place row LayerNorm (torch semantics: ddof=1, /(std+eps)).
// ---------------------------------------------------------------------------
__global__ void __launch_bounds__(256)
layernorm_kernel(float* __restrict__ out, const float* __restrict__ scale,
                 const float* __restrict__ offset) {
  const int row = blockIdx.x;
  float* xr = out + (size_t)row * D_MODEL;

  float x[4], s = 0.0f, s2 = 0.0f;
#pragma unroll
  for (int i = 0; i < 4; ++i) {
    x[i] = xr[threadIdx.x + i * 256];
    s  += x[i];
    s2 += x[i] * x[i];
  }
#pragma unroll
  for (int off = 16; off; off >>= 1) {
    s  += __shfl_xor(s,  off, 32);
    s2 += __shfl_xor(s2, off, 32);
  }
  __shared__ float ss[8], ss2[8];
  const int wv = threadIdx.x >> 5;
  if ((threadIdx.x & 31) == 0) { ss[wv] = s; ss2[wv] = s2; }
  __syncthreads();
  float ts = 0.0f, ts2 = 0.0f;
#pragma unroll
  for (int i = 0; i < 8; ++i) { ts += ss[i]; ts2 += ss2[i]; }

  const float mean = ts * (1.0f / 1024.0f);
  const float var  = (ts2 - 1024.0f * mean * mean) * (1.0f / 1023.0f);
  const float stdv = sqrtf(fmaxf(var, 0.0f));
  const float inv  = 1.0f / (stdv + 1e-5f);

#pragma unroll
  for (int i = 0; i < 4; ++i) {
    const int d = threadIdx.x + i * 256;
    xr[d] = scale[d] * (x[i] - mean) * inv + offset[d];
  }
}

// ---------------------------------------------------------------------------
extern "C" void kernel_launch(void* const* d_in, const int* in_sizes, int n_in,
                              void* d_out, int out_size, void* d_ws, size_t ws_size,
                              hipStream_t stream) {
  (void)in_sizes; (void)n_in; (void)out_size; (void)ws_size;
  const float* q      = (const float*)d_in[0];
  const float* k      = (const float*)d_in[1];
  const float* v      = (const float*)d_in[2];
  const float* wq     = (const float*)d_in[3];
  const float* wk     = (const float*)d_in[4];
  const float* wv     = (const float*)d_in[5];
  const float* wproj  = (const float*)d_in[6];
  const float* scale  = (const float*)d_in[7];
  const float* offset = (const float*)d_in[8];

  _Float16* ws = (_Float16*)d_ws;          // Qh | Kh | Vt | Ow  (f16)
  _Float16* Ow = ws + 3 * HEAD_ELEMS;
  float* out = (float*)d_out;

  qkv_proj_kernel<<<dim3(64, 16, 3), 128, 0, stream>>>(q, k, v, wq, wk, wv, ws);
  attention_kernel<<<dim3(64, 16, 4), 32, 0, stream>>>(ws, Ow);
  proj_residual_kernel<<<dim3(64, 16, 1), 128, 0, stream>>>(Ow, wproj, q, out);
  layernorm_kernel<<<dim3(4096, 1, 1), 256, 0, stream>>>(out, scale, offset);
}
